// EncoderBlock_33741263078130
// MI455X (gfx1250) — matrix operand order repaired and checked by reading, not compiled, not run on hardware
//
#include <hip/hip_runtime.h>
#include <hip/hip_bf16.h>
#include <math.h>

// ---------------------------------------------------------------------------
// EncoderBlock on MI455X (gfx1250): all GEMMs via v_wmma_f32_16x16x32_bf16,
// GEMM global->LDS staging via async-to-LDS DMA (ASYNCcnt), ping/pong
// buffers with a manually software-pipelined K loop (64 K / iteration).
// ---------------------------------------------------------------------------

#define Bz  4
#define Lz  2048
#define Dz  1024
#define NHz 16
#define KDz 64
#define MLPz 4096
#define Mz  (Bz*Lz)          // 8192 rows for all big GEMMs
#define EPSz 1e-5f

typedef __attribute__((ext_vector_type(16))) __bf16 v16bf;
typedef __attribute__((ext_vector_type(8)))  float  v8f;

union Frag {                 // one WMMA 16x32 bf16 operand (8 VGPRs / lane)
    v16bf v;
    unsigned int u[8];
};

#if __has_builtin(__builtin_amdgcn_s_wait_asynccnt)
#define ASYNC_WAIT0() __builtin_amdgcn_s_wait_asynccnt(0)
#else
#define ASYNC_WAIT0() asm volatile("s_wait_asynccnt 0" ::: "memory")
#endif

__device__ __forceinline__ unsigned short f2bf(float f) {
    union { float f; unsigned int u; } x; x.f = f;
    unsigned int r = x.u + 0x7FFFu + ((x.u >> 16) & 1u);   // round-nearest-even
    return (unsigned short)(r >> 16);
}

__device__ __forceinline__ float gelu_tanh(float v) {
    return 0.5f * v * (1.0f + tanhf(0.7978845608028654f * (v + 0.044715f * v * v * v)));
}

// async DMA of 16 bytes global -> LDS, tracked on ASYNCcnt (ISA 15.18.3 #98)
__device__ __forceinline__ void async_g2l_b128(const void* gptr, void* lptr) {
    unsigned int lds = (unsigned int)(size_t)lptr;   // flat LDS aperture: low
    asm volatile("global_load_async_to_lds_b128 %0, %1, off"
                 :: "v"(lds), "v"(gptr) : "memory");
}

// ---------------------------------------------------------------------------
// 1) scale = t_emb @ w_scale ; offset = t_emb @ w_offset   (tiny: 2x4096 dots)
// ---------------------------------------------------------------------------
__global__ __launch_bounds__(256) void scaleoff_kernel(
    const float* __restrict__ t_emb, const float* __restrict__ wsc,
    const float* __restrict__ wof, float* __restrict__ scale,
    float* __restrict__ offset)
{
    int t = blockIdx.x * 256 + threadIdx.x;          // 0 .. B*D
    int b = t / Dz, d = t % Dz;
    float s = 0.f, o = 0.f;
    for (int k = 0; k < Dz; ++k) {
        float te = t_emb[b * Dz + k];
        s += te * wsc[(size_t)k * Dz + d];
        o += te * wof[(size_t)k * Dz + d];
    }
    scale[t] = s; offset[t] = o;
}

// ---------------------------------------------------------------------------
// 2) LayerNorm variants -> bf16 output.
//    VAR==0 : y = ln(x)*scale[b,d] + offset[b,d]
//    VAR==1 : y = ln(x)*g[d]
// ---------------------------------------------------------------------------
template <int VAR>
__global__ __launch_bounds__(256) void ln_kernel(
    const float* __restrict__ x, const float* __restrict__ sc,
    const float* __restrict__ off, const float* __restrict__ g,
    unsigned short* __restrict__ out)
{
    __shared__ float rs[256], rs2[256];
    int row = blockIdx.x;                 // 0..M
    int b   = row / Lz;
    int tid = threadIdx.x;
    float s = 0.f, s2 = 0.f;
    #pragma unroll
    for (int i = 0; i < Dz / 256; ++i) {
        float v = x[(size_t)row * Dz + tid + i * 256];
        s += v; s2 += v * v;
    }
    rs[tid] = s; rs2[tid] = s2;
    __syncthreads();
    for (int st = 128; st > 0; st >>= 1) {
        if (tid < st) { rs[tid] += rs[tid + st]; rs2[tid] += rs2[tid + st]; }
        __syncthreads();
    }
    float mu  = rs[0] * (1.0f / Dz);
    float var = rs2[0] * (1.0f / Dz) - mu * mu;
    float rin = rsqrtf(var + EPSz);
    #pragma unroll
    for (int i = 0; i < Dz / 256; ++i) {
        int d = tid + i * 256;
        float v = (x[(size_t)row * Dz + d] - mu) * rin;
        if (VAR == 0) v = v * sc[b * Dz + d] + off[b * Dz + d];
        else          v = v * g[d];
        out[(size_t)row * Dz + d] = f2bf(v);
    }
}

// ---------------------------------------------------------------------------
// 3) fp32 [K][N] -> bf16 transposed [N][K]  (so WMMA B-fragments read
//    contiguous K-pairs per lane)
// ---------------------------------------------------------------------------
__global__ __launch_bounds__(256) void convT_kernel(
    const float* __restrict__ in, unsigned short* __restrict__ out,
    int Kd, int Nd)
{
    size_t t = (size_t)blockIdx.x * 256 + threadIdx.x;
    if (t >= (size_t)Kd * Nd) return;
    int n = (int)(t / Kd), k = (int)(t % Kd);
    out[t] = f2bf(in[(size_t)k * Nd + n]);
}

// ---------------------------------------------------------------------------
// 4) Tiled WMMA GEMM:  C[M,N] = A[M,K](bf16) * Bt[N,K](bf16)^T
//    Block tile 128x128, 8 waves, each wave 32x64 (8 accum frags).
//    Global->LDS via async DMA (ASYNCcnt); ping/pong buffers, manually
//    pipelined over 64 K per iteration (K % 64 == 0 for all uses), so each
//    loop body holds 16 statically-addressed WMMAs overlapping two DMAs.
//    EPI 0: Cf = val*alpha ; EPI 1: Cf = val + resid ; EPI 2: Cb = bf16(gelu)
// ---------------------------------------------------------------------------
template <int EPI>
__global__ __launch_bounds__(256) void gemm_bf16_kernel(
    const unsigned short* __restrict__ A,
    const unsigned short* __restrict__ Bt,
    float* __restrict__ Cf, unsigned short* __restrict__ Cb,
    const float* __restrict__ resid, float alpha, int N, int K)
{
    __shared__ __align__(16) unsigned short lA[2][128 * 32];
    __shared__ __align__(16) unsigned short lB[2][128 * 32];

    int nb  = N >> 7;
    int bx  = blockIdx.x % nb;
    int by  = blockIdx.x / nb;
    int tid = threadIdx.x;
    int wid = tid >> 5, lane = tid & 31;
    int hf  = lane >> 4, l15 = lane & 15;
    int wm  = wid & 3, wn = wid >> 2;      // 4x2 wave grid -> 128x128

    v8f acc[2][4];
    v8f vz = {};
    #pragma unroll
    for (int mf = 0; mf < 2; ++mf)
        #pragma unroll
        for (int nf = 0; nf < 4; ++nf) acc[mf][nf] = vz;

    const size_t Ar0 = (size_t)by * 128;
    const size_t Br0 = (size_t)bx * 128;

    // issue one 128x32 A tile + 128x32 Bt tile into buffer `buf` (async DMA)
    auto issue_tile = [&](int buf, int k0) {
        #pragma unroll
        for (int c = 0; c < 2; ++c) {
            int chunk = tid + c * 256;           // 0..511
            int row = chunk >> 2, seg = chunk & 3;
            async_g2l_b128(A  + (Ar0 + row) * K + k0 + seg * 8,
                           lA[buf] + row * 32 + seg * 8);
            async_g2l_b128(Bt + (Br0 + row) * K + k0 + seg * 8,
                           lB[buf] + row * 32 + seg * 8);
        }
    };

    // 8 WMMAs on one staged 32-K slice (buf is a compile-time constant)
    auto compute_tile = [&](int buf) {
        Frag fa[2], fb[4];
        #pragma unroll
        for (int mf = 0; mf < 2; ++mf) {
            int row = wm * 32 + mf * 16 + l15;
            #pragma unroll
            for (int v = 0; v < 8; ++v) {        // contiguous k = 16*hf + s (matches B)
                int k = hf * 16 + v * 2;
                fa[mf].u[v] = *(const unsigned int*)(lA[buf] + row * 32 + k);
            }
        }
        #pragma unroll
        for (int nf = 0; nf < 4; ++nf) {
            int rn = wn * 64 + nf * 16 + l15;
            #pragma unroll
            for (int v = 0; v < 8; ++v) {        // contiguous k = 16*hf + s (matches A)
                int k = hf * 16 + v * 2;
                fb[nf].u[v] = *(const unsigned int*)(lB[buf] + rn * 32 + k);
            }
        }
        #pragma unroll
        for (int mf = 0; mf < 2; ++mf)
            #pragma unroll
            for (int nf = 0; nf < 4; ++nf)
                acc[mf][nf] = __builtin_amdgcn_wmma_f32_16x16x32_bf16(
                    false, fa[mf].v, false, fb[nf].v,
                    (short)0, acc[mf][nf], false, false);
    };

    issue_tile(0, 0);
    ASYNC_WAIT0();
    __syncthreads();

    for (int k0 = 0; k0 < K; k0 += 64) {         // K % 64 == 0
        issue_tile(1, k0 + 32);                  // DMA overlaps 8 WMMAs
        compute_tile(0);
        ASYNC_WAIT0();
        __syncthreads();
        if (k0 + 64 < K) issue_tile(0, k0 + 64); // DMA overlaps 8 WMMAs
        compute_tile(1);
        ASYNC_WAIT0();
        __syncthreads();
    }

    // epilogue: C frag layout -> row = i + 8*half, col = lane&15
    #pragma unroll
    for (int mf = 0; mf < 2; ++mf)
        #pragma unroll
        for (int nf = 0; nf < 4; ++nf)
            #pragma unroll
            for (int i = 0; i < 8; ++i) {
                int row = by * 128 + wm * 32 + mf * 16 + i + hf * 8;
                int col = bx * 128 + wn * 64 + nf * 16 + l15;
                size_t idx = (size_t)row * N + col;
                float val = acc[mf][nf][i];
                if (EPI == 0)      Cf[idx] = val * alpha;
                else if (EPI == 1) Cf[idx] = val + resid[idx];
                else               Cb[idx] = f2bf(gelu_tanh(val));
            }
}

// ---------------------------------------------------------------------------
// 5) RoPE + head-pack:  q,k -> bf16 [B,NH,L,KD];  v -> bf16 transposed
//    [B,NH,KD,L]  (so the PV WMMA B-fragment reads contiguous kv pairs)
// ---------------------------------------------------------------------------
__global__ __launch_bounds__(256) void ropepack_kernel(
    const float* __restrict__ qf, const float* __restrict__ kf,
    const float* __restrict__ vf, unsigned short* __restrict__ qb,
    unsigned short* __restrict__ kb, unsigned short* __restrict__ vt)
{
    int t  = blockIdx.x * 256 + threadIdx.x;    // 0 .. B*NH*L
    int l  = t % Lz;
    int bh = t / Lz;                            // b*NH + h
    int b  = bh / NHz, h = bh % NHz;
    const size_t src = ((size_t)b * Lz + l) * Dz + h * KDz;
    unsigned short* qd = qb + ((size_t)bh * Lz + l) * KDz;
    unsigned short* kd = kb + ((size_t)bh * Lz + l) * KDz;
    #pragma unroll
    for (int i = 0; i < KDz / 2; ++i) {
        float freq = __powf(10000.f, -2.0f * (float)i / (float)KDz);
        float sn, cs; __sincosf((float)l * freq, &sn, &cs);
        float q1 = qf[src + 2 * i], q2 = qf[src + 2 * i + 1];
        qd[2 * i]     = f2bf(q1 * cs - q2 * sn);
        qd[2 * i + 1] = f2bf(q2 * cs + q1 * sn);
        float k1 = kf[src + 2 * i], k2 = kf[src + 2 * i + 1];
        kd[2 * i]     = f2bf(k1 * cs - k2 * sn);
        kd[2 * i + 1] = f2bf(k2 * cs + k1 * sn);
    }
    #pragma unroll
    for (int d = 0; d < KDz; ++d)
        vt[((size_t)bh * KDz + d) * Lz + l] = f2bf(vf[src + d]);
}

// ---------------------------------------------------------------------------
// 6) Flash attention: one wave per 16-row Q block. logits via chained WMMA
//    (KD=64 -> 2x k=32), online softmax with 16-lane shfl_xor reductions,
//    P staged via LDS (intra-wave DS ops are in-order) to become the next
//    A-fragment, then 4 WMMAs against V^T.  Output -> bf16 [B,L,D].
// ---------------------------------------------------------------------------
__global__ __launch_bounds__(128) void attn_kernel(
    const unsigned short* __restrict__ qb,
    const unsigned short* __restrict__ kb,
    const unsigned short* __restrict__ vt,
    unsigned short* __restrict__ yb)
{
    __shared__ __align__(16) unsigned short Pl[4 * 16 * 32];
    int wid = threadIdx.x >> 5, lane = threadIdx.x & 31;
    int hf = lane >> 4, l15 = lane & 15;
    int gw   = blockIdx.x * 4 + wid;
    int qblk = gw & (Lz / 16 - 1);
    int bh   = gw >> 7;                         // L/16 = 128
    unsigned short* Pw = Pl + wid * 16 * 32;

    // Q A-fragments (16 rows x 64, rope'd + pre-scaled by 1/sqrt(D))
    Frag aq0, aq1;
    const unsigned short* qbase = qb + ((size_t)bh * Lz + qblk * 16) * KDz;
    #pragma unroll
    for (int v = 0; v < 8; ++v) {
        int k = hf * 16 + v * 2;
        aq0.u[v] = *(const unsigned int*)(qbase + (size_t)l15 * KDz + k);
        aq1.u[v] = *(const unsigned int*)(qbase + (size_t)l15 * KDz + 32 + k);
    }

    v8f yacc[4]; v8f vz = {};
    #pragma unroll
    for (int nf = 0; nf < 4; ++nf) yacc[nf] = vz;
    float mrow[8], lrow[8];
    #pragma unroll
    for (int i = 0; i < 8; ++i) { mrow[i] = -1e30f; lrow[i] = 0.f; }

    for (int kv0 = 0; kv0 < Lz; kv0 += 32) {
        v8f s[2];
        #pragma unroll
        for (int j = 0; j < 2; ++j) {
            Frag bk0, bk1;
            int kvr = kv0 + j * 16 + l15;
            const unsigned short* kbase = kb + ((size_t)bh * Lz + kvr) * KDz;
            #pragma unroll
            for (int v = 0; v < 8; ++v) {
                int kk = hf * 16 + v * 2;
                bk0.u[v] = *(const unsigned int*)(kbase + kk);
                bk1.u[v] = *(const unsigned int*)(kbase + 32 + kk);
            }
            v8f tacc = {};
            tacc = __builtin_amdgcn_wmma_f32_16x16x32_bf16(
                false, aq0.v, false, bk0.v, (short)0, tacc, false, false);
            tacc = __builtin_amdgcn_wmma_f32_16x16x32_bf16(
                false, aq1.v, false, bk1.v, (short)0, tacc, false, false);
            s[j] = tacc;
        }

        // online softmax: rows live within one 16-lane half per VGPR index
        #pragma unroll
        for (int i = 0; i < 8; ++i) {
            float a = s[0][i], c = s[1][i];
            float mx = fmaxf(a, c);
            mx = fmaxf(mx, __shfl_xor(mx, 1, 32));
            mx = fmaxf(mx, __shfl_xor(mx, 2, 32));
            mx = fmaxf(mx, __shfl_xor(mx, 4, 32));
            mx = fmaxf(mx, __shfl_xor(mx, 8, 32));
            float mnew = fmaxf(mrow[i], mx);
            float corr = __expf(mrow[i] - mnew);
            float e0 = __expf(a - mnew), e1 = __expf(c - mnew);
            float rsum = e0 + e1;
            rsum += __shfl_xor(rsum, 1, 32);
            rsum += __shfl_xor(rsum, 2, 32);
            rsum += __shfl_xor(rsum, 4, 32);
            rsum += __shfl_xor(rsum, 8, 32);
            lrow[i] = lrow[i] * corr + rsum;
            mrow[i] = mnew;
            #pragma unroll
            for (int nf = 0; nf < 4; ++nf) yacc[nf][i] = yacc[nf][i] * corr;
            int prow = i + hf * 8;
            Pw[prow * 32 + l15]      = f2bf(e0);
            Pw[prow * 32 + 16 + l15] = f2bf(e1);
        }
        // DS ops are in-order per wave; keep the compiler from reordering and
        // make the dependency explicit:
        asm volatile("s_wait_dscnt 0" ::: "memory");

        Frag ap;
        #pragma unroll
        for (int v = 0; v < 8; ++v) {
            int k = hf * 16 + v * 2;
            ap.u[v] = *(const unsigned int*)(Pw + l15 * 32 + k);
        }
        #pragma unroll
        for (int nf = 0; nf < 4; ++nf) {
            Frag bv;
            int kdc = nf * 16 + l15;
            const unsigned short* vb =
                vt + ((size_t)bh * KDz + kdc) * Lz + kv0 + hf * 16;
            #pragma unroll
            for (int v = 0; v < 8; ++v)
                bv.u[v] = *(const unsigned int*)(vb + v * 2);
            yacc[nf] = __builtin_amdgcn_wmma_f32_16x16x32_bf16(
                false, ap.v, false, bv.v, (short)0, yacc[nf], false, false);
        }
    }

    int b = bh >> 4, h = bh & 15;
    #pragma unroll
    for (int nf = 0; nf < 4; ++nf)
        #pragma unroll
        for (int i = 0; i < 8; ++i) {
            int row = qblk * 16 + i + hf * 8;
            int col = h * KDz + nf * 16 + l15;
            yb[((size_t)b * Lz + row) * Dz + col] = f2bf(yacc[nf][i] / lrow[i]);
        }
}

// ---------------------------------------------------------------------------
// Host orchestration
// ---------------------------------------------------------------------------
extern "C" void kernel_launch(void* const* d_in, const int* in_sizes, int n_in,
                              void* d_out, int out_size, void* d_ws, size_t ws_size,
                              hipStream_t stream)
{
    (void)in_sizes; (void)n_in; (void)out_size; (void)ws_size;
    const float* x      = (const float*)d_in[0];
    const float* t_emb  = (const float*)d_in[1];
    const float* w_sc   = (const float*)d_in[2];
    const float* w_of   = (const float*)d_in[3];
    const float* wq     = (const float*)d_in[4];
    const float* wk     = (const float*)d_in[5];
    const float* wv     = (const float*)d_in[6];
    const float* wo     = (const float*)d_in[7];
    const float* mln_g  = (const float*)d_in[8];
    const float* w1     = (const float*)d_in[9];
    const float* w2     = (const float*)d_in[10];
    float* out = (float*)d_out;

    char* base = (char*)d_ws;
    size_t off = 0;
    auto alloc = [&](size_t bytes) -> char* {
        char* p = base + off;
        off = (off + bytes + 255) & ~(size_t)255;
        return p;
    };
    float* scale  = (float*)alloc(Bz * Dz * 4);
    float* offs   = (float*)alloc(Bz * Dz * 4);
    unsigned short* wqT = (unsigned short*)alloc((size_t)Dz * Dz * 2);
    unsigned short* wkT = (unsigned short*)alloc((size_t)Dz * Dz * 2);
    unsigned short* wvT = (unsigned short*)alloc((size_t)Dz * Dz * 2);
    unsigned short* woT = (unsigned short*)alloc((size_t)Dz * Dz * 2);
    unsigned short* w1T = (unsigned short*)alloc((size_t)Dz * MLPz * 2);
    unsigned short* w2T = (unsigned short*)alloc((size_t)MLPz * Dz * 2);
    unsigned short* hB  = (unsigned short*)alloc((size_t)Mz * Dz * 2);
    float* qf = (float*)alloc((size_t)Mz * Dz * 4);
    float* kf = (float*)alloc((size_t)Mz * Dz * 4);
    float* vf = (float*)alloc((size_t)Mz * Dz * 4);
    unsigned short* qb = (unsigned short*)alloc((size_t)Mz * Dz * 2);
    unsigned short* kb = (unsigned short*)alloc((size_t)Mz * Dz * 2);
    unsigned short* vt = (unsigned short*)alloc((size_t)Mz * Dz * 2);
    // reuse: qf/kf/vf dead after ropepack
    unsigned short* yB  = (unsigned short*)qf;                 // 16MB in 32MB
    unsigned short* gB  = (unsigned short*)kf;                 // 64MB in kf+vf
    unsigned short* h2B = hB;                                  // hB dead after QKV

    // 1) scale/offset projections
    scaleoff_kernel<<<(Bz * Dz) / 256, 256, 0, stream>>>(t_emb, w_sc, w_of, scale, offs);
    // 2) LN1 -> bf16 h
    ln_kernel<0><<<Mz, 256, 0, stream>>>(x, scale, offs, nullptr, hB);
    // 3) weight converts (bf16, transposed)
    convT_kernel<<<(Dz * Dz) / 256, 256, 0, stream>>>(wq, wqT, Dz, Dz);
    convT_kernel<<<(Dz * Dz) / 256, 256, 0, stream>>>(wk, wkT, Dz, Dz);
    convT_kernel<<<(Dz * Dz) / 256, 256, 0, stream>>>(wv, wvT, Dz, Dz);
    convT_kernel<<<(Dz * Dz) / 256, 256, 0, stream>>>(wo, woT, Dz, Dz);
    convT_kernel<<<(Dz * MLPz) / 256, 256, 0, stream>>>(w1, w1T, Dz, MLPz);
    convT_kernel<<<(MLPz * Dz) / 256, 256, 0, stream>>>(w2, w2T, MLPz, Dz);
    // 4) QKV GEMMs (q pre-scaled by 1/sqrt(D) = 1/32)
    const int gQKV = (Mz / 128) * (Dz / 128);
    gemm_bf16_kernel<0><<<gQKV, 256, 0, stream>>>(hB, wqT, qf, nullptr, nullptr, 0.03125f, Dz, Dz);
    gemm_bf16_kernel<0><<<gQKV, 256, 0, stream>>>(hB, wkT, kf, nullptr, nullptr, 1.0f, Dz, Dz);
    gemm_bf16_kernel<0><<<gQKV, 256, 0, stream>>>(hB, wvT, vf, nullptr, nullptr, 1.0f, Dz, Dz);
    // 5) RoPE + head pack (v transposed)
    ropepack_kernel<<<(Bz * NHz * Lz) / 256, 256, 0, stream>>>(qf, kf, vf, qb, kb, vt);
    // 6) flash attention -> bf16 y [B,L,D]
    attn_kernel<<<(Bz * NHz * (Lz / 16)) / 4, 128, 0, stream>>>(qb, kb, vt, yB);
    // 7) x2 = x + y @ wo   -> d_out (fp32)
    gemm_bf16_kernel<1><<<gQKV, 256, 0, stream>>>(yB, woT, out, nullptr, x, 1.0f, Dz, Dz);
    // 8) LN2 -> bf16 h2
    ln_kernel<1><<<Mz, 256, 0, stream>>>(out, nullptr, nullptr, mln_g, h2B);
    // 9) g = gelu(h2 @ w1) -> bf16
    gemm_bf16_kernel<2><<<(Mz / 128) * (MLPz / 128), 256, 0, stream>>>(h2B, w1T, nullptr, gB, nullptr, 1.0f, MLPz, Dz);
    // 10) out = out + g @ w2
    gemm_bf16_kernel<1><<<gQKV, 256, 0, stream>>>(gB, w2T, out, nullptr, out, 1.0f, Dz, MLPz);
}
